// EncoderLayer_11991548691281
// MI455X (gfx1250) — matrix operand order repaired and checked by reading, not compiled, not run on hardware
//
#include <hip/hip_runtime.h>
#include <hip/hip_bf16.h>

// ---------------------------------------------------------------------------
// Problem constants (from reference)
// ---------------------------------------------------------------------------
#define BB   8
#define NN   8192
#define DD_  512
#define HH   8
#define DK_  64
#define DFF_ 2048
#define T2   512          // 2 * TOP selected tokens per batch
#define BH   (BB * HH)

typedef __attribute__((ext_vector_type(16))) __bf16 v16bf;
typedef __attribute__((ext_vector_type(8)))  __bf16 v8bf;
typedef __attribute__((ext_vector_type(4)))  __bf16 v4bf;
typedef __attribute__((ext_vector_type(2)))  __bf16 v2bf;
typedef __attribute__((ext_vector_type(8)))  float  v8f;
typedef __attribute__((ext_vector_type(4)))  float  v4f;
typedef __attribute__((ext_vector_type(2)))  float  v2f;
typedef __attribute__((ext_vector_type(4)))  unsigned int u32x4;
typedef __attribute__((ext_vector_type(8)))  int i32x8;
typedef __attribute__((ext_vector_type(4)))  int i32x4;

#define TILE_M 64
#define TILE_N 32
#define TILE_K 32
#define LDSPAD 8          // bf16 row pitch = 40 elements = 80 B (16-B aligned)

#if defined(__has_builtin)
#if __has_builtin(__builtin_amdgcn_tensor_load_to_lds) && __has_builtin(__builtin_amdgcn_s_wait_tensorcnt)
#define USE_TDM 1
#endif
#endif
#ifndef USE_TDM
#define USE_TDM 0
#endif

// ---------------------------------------------------------------------------
// Top-k selection: per-batch bitonic sort of (desc-order float key | index)
// ---------------------------------------------------------------------------
__global__ void topk_kernel(const float* __restrict__ c,
                            const int* __restrict__ rnd,
                            int* __restrict__ sel)
{
    __shared__ unsigned long long keys[NN];   // 64 KB LDS (WGP has 320 KB)
    const int b = blockIdx.x;
    const int tid = threadIdx.x;

    for (int i = tid; i < NN; i += 1024) {
        unsigned u = __float_as_uint(c[(size_t)b * NN + i]);
        u = (u & 0x80000000u) ? ~u : (u | 0x80000000u);   // ascending map
        u = ~u;                                           // descending on c
        keys[i] = ((unsigned long long)u << 32) | (unsigned)i;
    }
    __syncthreads();

    for (int k = 2; k <= NN; k <<= 1) {
        for (int j = k >> 1; j > 0; j >>= 1) {
            for (int i = tid; i < NN; i += 1024) {
                int ixj = i ^ j;
                if (ixj > i) {
                    bool up = ((i & k) == 0);
                    unsigned long long a = keys[i], bqq = keys[ixj];
                    if (up ? (a > bqq) : (a < bqq)) { keys[i] = bqq; keys[ixj] = a; }
                }
            }
            __syncthreads();
        }
    }

    if (tid < 256) keys[tid] &= 0xFFFFFFFFull;   // winning indices
    __syncthreads();
    for (int k = 2; k <= 256; k <<= 1) {
        for (int j = k >> 1; j > 0; j >>= 1) {
            if (tid < 256) {
                int i = tid, ixj = i ^ j;
                if (ixj > i) {
                    bool up = ((i & k) == 0);
                    unsigned long long a = keys[i], bqq = keys[ixj];
                    if (up ? (a > bqq) : (a < bqq)) { keys[i] = bqq; keys[ixj] = a; }
                }
            }
            __syncthreads();
        }
    }
    if (tid < 256) {
        sel[b * T2 + tid]       = (int)(keys[tid] & 0xFFFFFFFFull);
        sel[b * T2 + 256 + tid] = rnd[b * 256 + tid];
    }
}

// ---------------------------------------------------------------------------
// Gather selected rows of x -> tbl (f32) + tbl16 (bf16); 128 thr, 1 v4f each
// ---------------------------------------------------------------------------
__global__ void gather_kernel(const float* __restrict__ x,
                              const int* __restrict__ sel,
                              float* __restrict__ tbl,
                              __bf16* __restrict__ tbl16)
{
    const int bj = blockIdx.x;
    const int b = bj >> 9;
    const int src = sel[bj];
    const v4f v = ((const v4f*)(x + ((size_t)b * NN + src) * DD_))[threadIdx.x];
    ((v4f*)(tbl + (size_t)bj * DD_))[threadIdx.x] = v;
    v4bf w;
    w[0] = (__bf16)v.x; w[1] = (__bf16)v.y; w[2] = (__bf16)v.z; w[3] = (__bf16)v.w;
    ((v4bf*)(tbl16 + (size_t)bj * DD_))[threadIdx.x] = w;
}

__global__ void scatter_kernel(const float* __restrict__ xbl,
                               const int* __restrict__ sel,
                               float* __restrict__ y)
{
    const int bj = blockIdx.x;
    const int b = bj >> 9;
    const int dst = sel[bj];
    ((v4f*)(y + ((size_t)b * NN + dst) * DD_))[threadIdx.x] =
        ((const v4f*)(xbl + (size_t)bj * DD_))[threadIdx.x];
}

// ---------------------------------------------------------------------------
// f32 -> bf16 weight conversion, vectorized (n multiple of 4)
// ---------------------------------------------------------------------------
__global__ void f2bf_kernel(const float* __restrict__ src,
                            __bf16* __restrict__ dst, int n4)
{
    int i = blockIdx.x * blockDim.x + threadIdx.x;
    int stride = gridDim.x * blockDim.x;
    for (; i < n4; i += stride) {
        v4f v = ((const v4f*)src)[i];
        v4bf w;
        w[0] = (__bf16)v.x; w[1] = (__bf16)v.y; w[2] = (__bf16)v.z; w[3] = (__bf16)v.w;
        ((v4bf*)dst)[i] = w;
    }
}

// ---------------------------------------------------------------------------
// LayerNorm (f32 in, bf16 out): 256 threads, one v2f per lane (D=512)
// ---------------------------------------------------------------------------
__global__ void layernorm_bf16_kernel(const float* __restrict__ X,
                                      const float* __restrict__ gma,
                                      const float* __restrict__ bta,
                                      __bf16* __restrict__ Y)
{
    __shared__ float red[256];
    const size_t row = blockIdx.x;
    const int tid = threadIdx.x;
    const v2f xv = ((const v2f*)(X + row * DD_))[tid];
    red[tid] = xv.x + xv.y; __syncthreads();
    for (int st = 128; st > 0; st >>= 1) {
        if (tid < st) red[tid] += red[tid + st];
        __syncthreads();
    }
    const float mu = red[0] * (1.0f / DD_);
    __syncthreads();
    float d0 = xv.x - mu, d1 = xv.y - mu;
    red[tid] = d0 * d0 + d1 * d1; __syncthreads();
    for (int st = 128; st > 0; st >>= 1) {
        if (tid < st) red[tid] += red[tid + st];
        __syncthreads();
    }
    const float rstd = rsqrtf(red[0] * (1.0f / DD_) + 1e-5f);
    const v2f gv = ((const v2f*)gma)[tid];
    const v2f bv = ((const v2f*)bta)[tid];
    v2bf o;
    o[0] = (__bf16)(d0 * rstd * gv.x + bv.x);
    o[1] = (__bf16)(d1 * rstd * gv.y + bv.y);
    ((v2bf*)(Y + row * DD_))[tid] = o;
}

// ---------------------------------------------------------------------------
// bf16 fragment loaders: contiguous per-lane runs -> ds_load_b128 pairs
// A (16x32, ISA 7.12.2): lane(g,m) needs K in [8g,8g+8) and [16+8g,16+8g+8)
// B stored TRANSPOSED in LDS (Bst[n][k]): lane needs K in [8g,8g+8) and [16+8g,16+8g+8)
// ---------------------------------------------------------------------------
__device__ __forceinline__ v16bf load_a_frag(const __bf16* rowp, int g)
{
    v8bf a0 = *(const v8bf*)(rowp + 8 * g);
    v8bf a1 = *(const v8bf*)(rowp + 16 + 8 * g);
    return __builtin_shufflevector(a0, a1, 0,1,2,3,4,5,6,7,8,9,10,11,12,13,14,15);
}
__device__ __forceinline__ v16bf load_b_frag(const __bf16* rowp, int g)
{
    v8bf b0 = *(const v8bf*)(rowp + 8 * g);
    v8bf b1 = *(const v8bf*)(rowp + 16 + 8 * g);
    return __builtin_shufflevector(b0, b1, 0,1,2,3,4,5,6,7,8,9,10,11,12,13,14,15);
}

// ---------------------------------------------------------------------------
// Generic tiled bf16 GEMM, f32 accumulate (v_wmma_f32_16x16x32_bf16).
// 256 threads = 8 waves (4 M x 2 N), block tile 64x32, K-chunk 32.
// A tile staged by the Tensor Data Mover (strided 64x32 DMA with LDS row
// padding -> 80B pitch); B tile gathered transposed.
// ---------------------------------------------------------------------------
__global__ void gemm_bf16_kernel(const __bf16* __restrict__ A, int lda,
                                 const __bf16* __restrict__ Bw, int ldb,
                                 float* __restrict__ Cf, __bf16* __restrict__ Cb,
                                 int ldc, int M, int N, int K,
                                 const float* __restrict__ bias,
                                 const float* __restrict__ Res, int ldr,
                                 float alpha, int relu)
{
    __shared__ __align__(16) __bf16 As[TILE_M][TILE_K + LDSPAD];
    __shared__ __align__(16) __bf16 Bst[TILE_N][TILE_K + LDSPAD];
    const int tid  = threadIdx.x;
    const int wave = tid >> 5, lane = tid & 31;
    const int wm = wave & 3, wn = wave >> 2;
    const int g = lane >> 4, mn = lane & 15;
    const int rowBase = blockIdx.y * TILE_M;
    const int colBase = blockIdx.x * TILE_N;

    v8f acc = {};
    for (int k0 = 0; k0 < K; k0 += TILE_K) {
#if USE_TDM
        if (tid < 32) {
            // --- Tensor DMA descriptor (ISA 8.3/8.4): 2-D tile 32x64 bf16 ---
            const unsigned long long ga =
                (unsigned long long)(size_t)A + ((size_t)rowBase * lda + k0) * 2ull;
            const unsigned ldsa = (unsigned)(size_t)&As[0][0];
            u32x4 g0;
            g0[0] = 1u;                                   // count=1, user mode
            g0[1] = ldsa;                                 // lds_addr
            g0[2] = (unsigned)(ga & 0xFFFFFFFFull);       // global_addr[31:0]
            g0[3] = (unsigned)((ga >> 32) & 0x1FFFFFFull) // global_addr[56:32]
                  | (2u << 30);                           // type = image
            i32x8 g1;
            g1[0] = (int)((1u << 16)      // data_size = 2 bytes
                  | (1u << 20)            // pad_enable
                  | (3u << 22)            // pad_interval: every 16 DWORDs (one row)
                  | (3u << 25));          // pad_amount: 4 DWORDs (+16 B -> 80 B pitch)
            g1[1] = (int)(((unsigned)lda & 0xFFFFu) << 16);           // tensor_dim0 lo
            g1[2] = (int)((((unsigned)lda >> 16) & 0xFFFFu)
                  | (((unsigned)M & 0xFFFFu) << 16));                 // dim0 hi | dim1 lo
            g1[3] = (int)((((unsigned)M >> 16) & 0xFFFFu)
                  | ((unsigned)TILE_K << 16));                        // dim1 hi | tile_dim0
            g1[4] = (int)TILE_M;                                      // tile_dim1
            g1[5] = (int)lda;                                         // dim0_stride lo
            g1[6] = 0;                                                // dim0_stride hi
            g1[7] = 0;
            i32x4 gz4 = {};
            i32x8 gz8 = {};
            // clang-23 / therock form: 6 args (g0, g1, g2, g3, extra group, cpol)
            __builtin_amdgcn_tensor_load_to_lds(g0, g1, gz4, gz4, gz8, 0);
        }
#else
        {   // fallback: one 16-B load per thread (64 rows x 4 chunks)
            const int r = tid >> 2, q = (tid & 3) * 8;
            *(u32x4*)&As[r][q] =
                *(const u32x4*)&A[(size_t)(rowBase + r) * lda + k0 + q];
        }
#endif
        {   // B tile, transposed into LDS: 4-element column gather + b64 store
            const int n = tid & 31, ks = (tid >> 5) * 4;
            v4bf t;
            #pragma unroll
            for (int i2 = 0; i2 < 4; ++i2)
                t[i2] = Bw[(size_t)(k0 + ks + i2) * ldb + colBase + n];
            *(v4bf*)&Bst[n][ks] = t;
        }
        if (k0 + TILE_K < K)
            __builtin_prefetch(&Bw[(size_t)(k0 + TILE_K + (tid >> 5)) * ldb + colBase], 0, 1);
#if USE_TDM
        if (tid < 32) __builtin_amdgcn_s_wait_tensorcnt(0);
#endif
        __syncthreads();

        v16bf af = load_a_frag(&As[wm * 16 + mn][0], g);
        v16bf bfrag = load_b_frag(&Bst[wn * 16 + mn][0], g);
        acc = __builtin_amdgcn_wmma_f32_16x16x32_bf16(
            false, af, false, bfrag, (short)0, acc, false, false);
        __syncthreads();
    }

    const int col = colBase + wn * 16 + mn;
    const float bsv = bias ? bias[col] : 0.0f;
    #pragma unroll
    for (int r = 0; r < 8; ++r) {
        int row = rowBase + wm * 16 + g * 8 + r;   // C/D: lanes 16-31 => M+8
        float v = alpha * acc[r] + bsv;
        if (Res)  v += Res[(size_t)row * ldr + col];
        if (relu) v = fmaxf(v, 0.0f);
        if (Cb) Cb[(size_t)row * ldc + col] = (__bf16)v;
        else    Cf[(size_t)row * ldc + col] = v;
    }
}

// ---------------------------------------------------------------------------
// Attention scores: S[b,h,q,n] = 0.125 * sum_d Q[b,q,h,d]*K[b,n,h,d]
// per (b,h): M=512, N=8192, K=64. K rows are contiguous along d, so BOTH
// tiles stage with 16-B vector loads (B lands naturally transposed).
// ---------------------------------------------------------------------------
__global__ void attn_scores_kernel(const __bf16* __restrict__ Q,
                                   const __bf16* __restrict__ Kt,
                                   float* __restrict__ S)
{
    __shared__ __align__(16) __bf16 As[TILE_M][TILE_K + LDSPAD];
    __shared__ __align__(16) __bf16 Bst[TILE_N][TILE_K + LDSPAD];
    const int bh = blockIdx.z, b = bh >> 3, h = bh & 7;
    const __bf16* Ab = Q  + (size_t)b * T2 * DD_ + h * DK_;
    const __bf16* Kb = Kt + (size_t)b * NN * DD_ + h * DK_;
    float* Sb = S + (size_t)bh * T2 * NN;

    const int tid  = threadIdx.x;
    const int wave = tid >> 5, lane = tid & 31;
    const int wm = wave & 3, wn = wave >> 2;
    const int g = lane >> 4, mn = lane & 15;
    const int rowBase = blockIdx.y * TILE_M;
    const int colBase = blockIdx.x * TILE_N;

    v8f acc = {};
    #pragma unroll
    for (int k0 = 0; k0 < DK_; k0 += TILE_K) {
        {   const int r = tid >> 2, q = (tid & 3) * 8;
            *(u32x4*)&As[r][q] = *(const u32x4*)&Ab[(size_t)(rowBase + r) * DD_ + k0 + q];
        }
        if (tid < 128) {   // 32 key-rows x 4 chunks (wave-uniform branch)
            const int n = tid >> 2, q = (tid & 3) * 8;
            *(u32x4*)&Bst[n][q] = *(const u32x4*)&Kb[(size_t)(colBase + n) * DD_ + k0 + q];
        }
        __syncthreads();
        v16bf af = load_a_frag(&As[wm * 16 + mn][0], g);
        v16bf bfrag = load_b_frag(&Bst[wn * 16 + mn][0], g);
        acc = __builtin_amdgcn_wmma_f32_16x16x32_bf16(
            false, af, false, bfrag, (short)0, acc, false, false);
        __syncthreads();
    }
    const int col = colBase + wn * 16 + mn;
    #pragma unroll
    for (int r = 0; r < 8; ++r) {
        int row = rowBase + wm * 16 + g * 8 + r;
        Sb[(size_t)row * NN + col] = 0.125f * acc[r];   // 1/sqrt(64)
    }
}

// ---------------------------------------------------------------------------
// Row softmax over 8192 columns, in place, v4f vectorized
// ---------------------------------------------------------------------------
__global__ void softmax_kernel(float* __restrict__ P)
{
    __shared__ float red[256];
    v4f* p4 = (v4f*)(P + (size_t)blockIdx.x * NN);
    const int tid = threadIdx.x;
    float m = -3.402823466e+38f;
    for (int i = tid; i < NN / 4; i += 256) {
        v4f v = p4[i];
        m = fmaxf(m, fmaxf(fmaxf(v.x, v.y), fmaxf(v.z, v.w)));
    }
    red[tid] = m; __syncthreads();
    for (int st = 128; st > 0; st >>= 1) {
        if (tid < st) red[tid] = fmaxf(red[tid], red[tid + st]);
        __syncthreads();
    }
    m = red[0];
    __syncthreads();
    float sum = 0.f;
    for (int i = tid; i < NN / 4; i += 256) {
        v4f v = p4[i];
        v.x = __expf(v.x - m); v.y = __expf(v.y - m);
        v.z = __expf(v.z - m); v.w = __expf(v.w - m);
        p4[i] = v;
        sum += v.x + v.y + v.z + v.w;
    }
    red[tid] = sum; __syncthreads();
    for (int st = 128; st > 0; st >>= 1) {
        if (tid < st) red[tid] += red[tid + st];
        __syncthreads();
    }
    const float inv = 1.0f / red[0];
    for (int i = tid; i < NN / 4; i += 256) {
        v4f v = p4[i];
        v.x *= inv; v.y *= inv; v.z *= inv; v.w *= inv;
        p4[i] = v;
    }
}

// ---------------------------------------------------------------------------
// O = P @ V per (b,h): f32 WMMA 16x16x4, M=512, N=64, K=8192.
// A rows contiguous (v4f loads); V columns gathered transposed (v2f stores).
// ---------------------------------------------------------------------------
__global__ void attn_pv_kernel(const float* __restrict__ P,
                               const float* __restrict__ V,
                               __bf16* __restrict__ O)
{
    __shared__ __align__(16) float As[64][20];    // 80-B pitch
    __shared__ __align__(16) float Bst[32][20];
    const int bh = blockIdx.z, b = bh >> 3, h = bh & 7;
    const float* Ab = P + (size_t)bh * T2 * NN;
    const float* Vb = V + (size_t)b * NN * DD_ + h * DK_;
    __bf16* Ob = O + (size_t)b * T2 * DD_ + h * DK_;

    const int tid  = threadIdx.x;
    const int wave = tid >> 5, lane = tid & 31;
    const int wm = wave & 3, wn = wave >> 2;
    const int g = lane >> 4, mn = lane & 15;
    const int rowBase = blockIdx.y * 64;
    const int colBase = blockIdx.x * 32;

    v8f acc = {};
    for (int k0 = 0; k0 < NN; k0 += 16) {
        {   const int r = tid >> 2, q = (tid & 3) * 4;      // 64 rows x 4 v4f
            *(v4f*)&As[r][q] = *(const v4f*)&Ab[(size_t)(rowBase + r) * NN + k0 + q];
        }
        {   const int n = tid & 31, ks = (tid >> 5) * 2;    // V columns -> Bst[n][k]
            v2f t;
            t.x = Vb[(size_t)(k0 + ks)     * DD_ + colBase + n];
            t.y = Vb[(size_t)(k0 + ks + 1) * DD_ + colBase + n];
            *(v2f*)&Bst[n][ks] = t;
        }
        if (k0 + 16 < NN)
            __builtin_prefetch(&Ab[(size_t)(rowBase + (tid >> 2)) * NN + k0 + 16], 0, 1);
        __syncthreads();
        #pragma unroll
        for (int kk = 0; kk < 16; kk += 4) {
            // ISA 7.12.2: 32-bit A 16x4 — lanes 0-15 K={0,1}, lanes 16-31 K={2,3}
            v2f a  = *(const v2f*)&As[wm * 16 + mn][kk + 2 * g];
            v2f bb = *(const v2f*)&Bst[wn * 16 + mn][kk + 2 * g];
            acc = __builtin_amdgcn_wmma_f32_16x16x4_f32(
                false, a, false, bb, (short)0, acc, false, false);
        }
        __syncthreads();
    }
    const int col = colBase + wn * 16 + mn;
    #pragma unroll
    for (int r = 0; r < 8; ++r) {
        int row = rowBase + wm * 16 + g * 8 + r;
        Ob[(size_t)row * DD_ + col] = (__bf16)acc[r];
    }
}

// ---------------------------------------------------------------------------
// Host-side orchestration
// ---------------------------------------------------------------------------
extern "C" void kernel_launch(void* const* d_in, const int* in_sizes, int n_in,
                              void* d_out, int out_size, void* d_ws, size_t ws_size,
                              hipStream_t stream)
{
    (void)in_sizes; (void)n_in; (void)out_size; (void)ws_size;

    const float* x   = (const float*)d_in[0];
    const float* c   = (const float*)d_in[1];
    const float* g1  = (const float*)d_in[2];
    const float* b1  = (const float*)d_in[3];
    const float* g2  = (const float*)d_in[4];
    const float* b2  = (const float*)d_in[5];
    const float* Wq  = (const float*)d_in[6];
    const float* bq  = (const float*)d_in[7];
    const float* Wk  = (const float*)d_in[8];
    const float* bk  = (const float*)d_in[9];
    const float* Wv  = (const float*)d_in[10];
    const float* bvp = (const float*)d_in[11];
    const float* Wo  = (const float*)d_in[12];
    const float* bo  = (const float*)d_in[13];
    const float* W1  = (const float*)d_in[14];
    const float* bf1 = (const float*)d_in[15];
    const float* W2  = (const float*)d_in[16];
    const float* bf2 = (const float*)d_in[17];
    const int*   rnd = (const int*)d_in[18];

    const size_t BND = (size_t)BB * NN * DD_;
    const size_t T2D = (size_t)BB * T2 * DD_;
    const size_t BNF = (size_t)BB * NN * DFF_;
    const size_t DDl = (size_t)DD_ * DD_;
    const size_t DFD = (size_t)DD_ * DFF_;

    char* p = (char*)d_ws;
    auto alloc = [&](size_t bytes) -> void* {
        void* q = (void*)p;
        p += (bytes + 255) & ~(size_t)255;
        return q;
    };
    float*  vb    = (float*)alloc(BND * 4);
    float*  yb    = (float*)alloc(BND * 4);
    float*  tbl   = (float*)alloc(T2D * 4);
    float*  xbl   = (float*)alloc(T2D * 4);
    __bf16* xn16  = (__bf16*)alloc(BND * 2);
    __bf16* yn16  = (__bf16*)alloc(BND * 2);
    __bf16* kb16  = (__bf16*)alloc(BND * 2);
    __bf16* qb16  = (__bf16*)alloc(T2D * 2);
    __bf16* ob16  = (__bf16*)alloc(T2D * 2);
    __bf16* tbl16 = (__bf16*)alloc(T2D * 2);
    __bf16* h116  = (__bf16*)alloc(BNF * 2);
    __bf16* Wq16  = (__bf16*)alloc(DDl * 2);
    __bf16* Wk16  = (__bf16*)alloc(DDl * 2);
    __bf16* Wv16  = (__bf16*)alloc(DDl * 2);
    __bf16* Wo16  = (__bf16*)alloc(DDl * 2);
    __bf16* W116  = (__bf16*)alloc(DFD * 2);
    __bf16* W216  = (__bf16*)alloc(DFD * 2);
    int*    sel   = (int*)alloc((size_t)BB * T2 * 4);

    float* out  = (float*)d_out;
    float* attn = out + BND;   // [B,H,2T,N]

    topk_kernel<<<BB, 1024, 0, stream>>>(c, rnd, sel);
    gather_kernel<<<BB * T2, 128, 0, stream>>>(x, sel, tbl, tbl16);
    layernorm_bf16_kernel<<<BB * NN, 256, 0, stream>>>(x, g1, b1, xn16);
    f2bf_kernel<<<128, 256, 0, stream>>>(Wq, Wq16, (int)(DDl / 4));
    f2bf_kernel<<<128, 256, 0, stream>>>(Wk, Wk16, (int)(DDl / 4));
    f2bf_kernel<<<128, 256, 0, stream>>>(Wv, Wv16, (int)(DDl / 4));
    f2bf_kernel<<<128, 256, 0, stream>>>(Wo, Wo16, (int)(DDl / 4));
    f2bf_kernel<<<256, 256, 0, stream>>>(W1, W116, (int)(DFD / 4));
    f2bf_kernel<<<256, 256, 0, stream>>>(W2, W216, (int)(DFD / 4));

    // projections (bf16 WMMA, f32 accumulate)
    gemm_bf16_kernel<<<dim3(DD_ / TILE_N, (BB * NN) / TILE_M), 256, 0, stream>>>(
        xn16, DD_, Wk16, DD_, nullptr, kb16, DD_,
        BB * NN, DD_, DD_, bk, nullptr, 0, 1.0f, 0);
    gemm_bf16_kernel<<<dim3(DD_ / TILE_N, (BB * NN) / TILE_M), 256, 0, stream>>>(
        xn16, DD_, Wv16, DD_, vb, nullptr, DD_,
        BB * NN, DD_, DD_, bvp, nullptr, 0, 1.0f, 0);
    gemm_bf16_kernel<<<dim3(DD_ / TILE_N, (BB * T2) / TILE_M), 256, 0, stream>>>(
        tbl16, DD_, Wq16, DD_, nullptr, qb16, DD_,
        BB * T2, DD_, DD_, bq, nullptr, 0, 1.0f, 0);

    // attention
    attn_scores_kernel<<<dim3(NN / TILE_N, T2 / TILE_M, BH), 256, 0, stream>>>(
        qb16, kb16, attn);
    softmax_kernel<<<BH * T2, 256, 0, stream>>>(attn);
    attn_pv_kernel<<<dim3(DK_ / 32, T2 / 64, BH), 256, 0, stream>>>(attn, vb, ob16);

    // o @ Wo + bo + top_big_lambda -> xbl
    gemm_bf16_kernel<<<dim3(DD_ / TILE_N, (BB * T2) / TILE_M), 256, 0, stream>>>(
        ob16, DD_, Wo16, DD_, xbl, nullptr, DD_,
        BB * T2, DD_, DD_, bo, tbl, DD_, 1.0f, 0);

    // scatter back: y = x; y[sel] = xbl
    (void)hipMemcpyAsync(yb, x, BND * sizeof(float), hipMemcpyDeviceToDevice, stream);
    scatter_kernel<<<BB * T2, 128, 0, stream>>>(xbl, sel, yb);

    // FFN
    layernorm_bf16_kernel<<<BB * NN, 256, 0, stream>>>(yb, g2, b2, yn16);
    gemm_bf16_kernel<<<dim3(DFF_ / TILE_N, (BB * NN) / TILE_M), 256, 0, stream>>>(
        yn16, DD_, W116, DFF_, nullptr, h116, DFF_,
        BB * NN, DFF_, DD_, bf1, nullptr, 0, 1.0f, 1);
    gemm_bf16_kernel<<<dim3(DD_ / TILE_N, (BB * NN) / TILE_M), 256, 0, stream>>>(
        h116, DFF_, W216, DD_, out, nullptr, DD_,
        BB * NN, DD_, DFF_, bf2, yb, DD_, 1.0f, 0);
}
